// GraphHMSJepa_36026185679474
// MI455X (gfx1250) — compile-verified, matrix operand order checked by reading, not run on hardware
//
#include <hip/hip_runtime.h>

// ---------------- problem constants (from reference) ----------------
#define B_     128
#define NP0_   32
#define NP1_   8
#define NP2_   2
#define H_     128
#define P0_    (B_*NP0_)   // 4096
#define P1_    (B_*NP1_)   // 1024
#define P2_    (B_*NP2_)   // 256
#define NNODE_ 200000
#define NEDGE_ 400000
#define NSUB_  500000
#define ESUB_  1500000
#define FNODE_ 64
#define FEDGE_ 16
#define RW_    16
#define NHEAD_ 4
#define DH_    32
#define NL_GNN 4

typedef __attribute__((ext_vector_type(16))) _Float16 v16h;
typedef __attribute__((ext_vector_type(2)))  _Float16 h2;
typedef __attribute__((ext_vector_type(8)))  float    v8f;

static inline int ceil_div(int a, int b) { return (a + b - 1) / b; }

// =====================================================================
// WMMA GEMM:  C[M,N] = epilogue( (A[g(m),K] (+A2[m,K])) @ W[K,N] + bias )
// epilogue: optional ReLU, then optional residual add (with row gather).
// Block = 256 threads = 8 waves: 64 rows (4 m-tiles) x 128 cols (8 wave
// n-tiles). A tile and W slab staged in LDS as f16; W is stored
// TRANSPOSED ([n][k]) so both A and B fragments are two contiguous
// 16-byte chunks per lane -> 2x ds_load_b128 each. Row pitch = 40 halves
// (80 B): multiple of 16 B (b128-legal) and 20-bank stride with
// gcd(20,64)=4 so 16 consecutive rows tile all 64 banks conflict-free.
// MACs: v_wmma_f32_16x16x32_f16 (f32 accumulate).
// =====================================================================
#define MT_ 4
#define PITCH_ 40
__global__ __launch_bounds__(256) void wmma_gemm_kernel(
    const float* __restrict__ A, const float* __restrict__ A2,
    const int* __restrict__ gidx,
    const float* __restrict__ W, const float* __restrict__ bias,
    const float* __restrict__ resid, const int* __restrict__ ridx,
    float* __restrict__ C, int M, int N, int K, int act)
{
    __shared__ _Float16 sA[MT_ * 16][PITCH_];   // 64 rows x 32 used halves
    __shared__ _Float16 sW[128][PITCH_];        // transposed: [n][k]

    const int tid  = threadIdx.x;
    const int wv   = tid >> 5;
    const int lane = tid & 31;
    const int lm   = lane & 15;
    const int lh   = lane >> 4;
    const int m0   = blockIdx.x * (MT_ * 16);
    const int nb0  = blockIdx.y * 128;
    const int n0   = nb0 + wv * 16;

    v8f acc[MT_];
#pragma unroll
    for (int t = 0; t < MT_; ++t) acc[t] = (v8f){0.f,0.f,0.f,0.f,0.f,0.f,0.f,0.f};

    for (int k0 = 0; k0 < K; k0 += 32) {
        // ---- stage A tile: 64 rows x 16 k-pairs, packed 2xf16 stores ----
        for (int e = tid; e < MT_ * 16 * 16; e += 256) {
            int r  = e >> 4;              // row in tile
            int cp = (e & 15) << 1;       // k pair base
            int row = m0 + r;
            float x0 = 0.f, x1 = 0.f;
            if (row < M) {
                int ar = gidx ? gidx[row] : row;
                const float* ap = A + (size_t)ar * K;
                int kk = k0 + cp;
                if (kk < K)     x0 = ap[kk];
                if (kk + 1 < K) x1 = ap[kk + 1];
                if (A2) {
                    const float* a2p = A2 + (size_t)row * K;
                    if (kk < K)     x0 += a2p[kk];
                    if (kk + 1 < K) x1 += a2p[kk + 1];
                }
            }
            *(h2*)&sA[r][cp] = (h2){(_Float16)x0, (_Float16)x1};
        }
        // ---- stage W slab transposed: 128 cols x 16 k-pairs ----
        for (int e = tid; e < 128 * 16; e += 256) {
            int n  = e & 127;             // consecutive threads -> consecutive n (coalesced reads)
            int kp = (e >> 7) << 1;
            int nn = nb0 + n;
            float w0 = 0.f, w1 = 0.f;
            if (nn < N) {
                int kk = k0 + kp;
                if (kk < K)     w0 = W[(size_t)kk * N + nn];
                if (kk + 1 < K) w1 = W[(size_t)(kk + 1) * N + nn];
            }
            *(h2*)&sW[n][kp] = (h2){(_Float16)w0, (_Float16)w1};
        }
        __syncthreads();

        if (n0 < N) {
            // B fragment: lane (n=lm, half=lh); k-order per ISA 16-bit layout.
            // Halves {8*lh..8*lh+7} then {16+8*lh..16+8*lh+7}: two contiguous
            // 16-byte chunks -> 2x ds_load_b128.
            v16h bf;
#pragma unroll
            for (int p = 0; p < 8; ++p) {
                int kk = ((p & 3) << 1) + ((p >> 2) << 4) + (lh << 3);
                bf[2 * p]     = sW[wv * 16 + lm][kk];
                bf[2 * p + 1] = sW[wv * 16 + lm][kk + 1];
            }
#pragma unroll
            for (int t = 0; t < MT_; ++t) {
                v16h af;
#pragma unroll
                for (int p = 0; p < 8; ++p) {
                    int kk = ((p & 3) << 1) + ((p >> 2) << 4) + (lh << 3);
                    af[2 * p]     = sA[t * 16 + lm][kk];
                    af[2 * p + 1] = sA[t * 16 + lm][kk + 1];
                }
                acc[t] = __builtin_amdgcn_wmma_f32_16x16x32_f16(
                    false, af, false, bf, (short)0, acc[t], false, false);
            }
        }
        __syncthreads();
    }

    if (n0 < N) {
        int n = n0 + lm;
        float bv = (bias && n < N) ? bias[n] : 0.f;
#pragma unroll
        for (int t = 0; t < MT_; ++t) {
#pragma unroll
            for (int r = 0; r < 8; ++r) {
                int m = m0 + t * 16 + r + 8 * lh;   // C/D layout: VGPR r, half lh
                if (m < M && n < N) {
                    float xv = acc[t][r] + bv;
                    if (act) xv = fmaxf(xv, 0.f);
                    if (resid) {
                        int rm = ridx ? ridx[m] : m;
                        xv += resid[(size_t)rm * N + n];
                    }
                    C[(size_t)m * N + n] = xv;
                }
            }
        }
    }
}

// ---------------- elementwise / scatter-gather kernels ----------------
__global__ void zero_kernel(float* p, int n) {
    int t = blockIdx.x * blockDim.x + threadIdx.x;
    if (t < n) p[t] = 0.f;
}

__global__ void gather_rows_kernel(float* __restrict__ out, const float* __restrict__ in,
                                   const int* __restrict__ idx, int Nrow, int F) {
    int t = blockIdx.x * blockDim.x + threadIdx.x;
    if (t >= Nrow * F) return;
    int i = t / F, c = t - i * F;
    out[(size_t)i * F + c] = in[(size_t)idx[i] * F + c];
}

__global__ void add_gather_kernel(float* __restrict__ x, const float* __restrict__ u,
                                  const int* __restrict__ idx, int Nrow, int F) {
    int t = blockIdx.x * blockDim.x + threadIdx.x;
    if (t >= Nrow * F) return;
    int i = t / F, c = t - i * F;
    x[(size_t)i * F + c] += u[(size_t)idx[i] * F + c];
}

__global__ void seg_accum_kernel(const float* __restrict__ x, const int* __restrict__ idx,
                                 float* __restrict__ sum, float* __restrict__ cnt,
                                 int Nrow, int F) {
    int t = blockIdx.x * blockDim.x + threadIdx.x;
    if (t >= Nrow * F) return;
    int i = t / F, c = t - i * F;
    int p = idx[i];
    atomicAdd(&sum[(size_t)p * F + c], x[(size_t)i * F + c]);
    if (c == 0) atomicAdd(&cnt[p], 1.f);
}

__global__ void seg_div_kernel(float* __restrict__ sum, const float* __restrict__ cnt,
                               int P, int F) {
    int t = blockIdx.x * blockDim.x + threadIdx.x;
    if (t >= P * F) return;
    sum[t] /= fmaxf(cnt[t / F], 1.f);
}

// raw0 = seg_max(rw[nmap], batch, P0); inputs are in [0,1) so uint-bitwise
// atomicMax == float max, and zero-init matches where(isfinite, m, 0).
__global__ void segmax_gather_kernel(const float* __restrict__ rw, const int* __restrict__ nmap,
                                     const int* __restrict__ batch, float* __restrict__ out,
                                     int Nrow) {
    int t = blockIdx.x * blockDim.x + threadIdx.x;
    if (t >= Nrow * RW_) return;
    int i = t / RW_, c = t - i * RW_;
    float v = rw[(size_t)nmap[i] * RW_ + c];
    atomicMax((unsigned int*)&out[(size_t)batch[i] * RW_ + c], __float_as_uint(v));
}

// GINE message + aggregate: wave per edge, float4 per lane (32*4 = 128).
__global__ __launch_bounds__(256) void gine_msg_kernel(
    const float* __restrict__ x, const float* __restrict__ eenc,
    const int* __restrict__ ei, const int* __restrict__ emap,
    float* __restrict__ agg, int E)
{
    int j = blockIdx.x * 8 + (threadIdx.x >> 5);
    int lane = threadIdx.x & 31;
    if (j >= E) return;
    int src = ei[j], dst = ei[E + j], em = emap[j];
    float4 xv = ((const float4*)(x + (size_t)src * H_))[lane];
    float4 ev = ((const float4*)(eenc + (size_t)em * H_))[lane];
    float4 mv;
    mv.x = fmaxf(xv.x + ev.x, 0.f);
    mv.y = fmaxf(xv.y + ev.y, 0.f);
    mv.z = fmaxf(xv.z + ev.z, 0.f);
    mv.w = fmaxf(xv.w + ev.w, 0.f);
    float* ap = agg + (size_t)dst * H_ + lane * 4;
    atomicAdd(ap + 0, mv.x);
    atomicAdd(ap + 1, mv.y);
    atomicAdd(ap + 2, mv.z);
    atomicAdd(ap + 3, mv.w);
}

// LayerNorm over H=128: one wave32 per row, 4 elems/lane, shfl reductions.
__global__ void ln_kernel(const float* __restrict__ x, const float* __restrict__ g,
                          const float* __restrict__ b, float* __restrict__ y, int M) {
    int row = blockIdx.x * (blockDim.x >> 5) + (threadIdx.x >> 5);
    int lane = threadIdx.x & 31;
    if (row >= M) return;
    float4 v = ((const float4*)(x + (size_t)row * H_))[lane];
    float s = v.x + v.y + v.z + v.w;
    for (int o = 16; o > 0; o >>= 1) s += __shfl_xor(s, o, 32);
    float mean = s * (1.f / (float)H_);
    float d0 = v.x - mean, d1 = v.y - mean, d2 = v.z - mean, d3 = v.w - mean;
    float q = d0 * d0 + d1 * d1 + d2 * d2 + d3 * d3;
    for (int o = 16; o > 0; o >>= 1) q += __shfl_xor(q, o, 32);
    float inv = rsqrtf(q * (1.f / (float)H_) + 1e-5f);
    float4 gv = ((const float4*)g)[lane];
    float4 bv = ((const float4*)b)[lane];
    float4 o4;
    o4.x = d0 * inv * gv.x + bv.x;
    o4.y = d1 * inv * gv.y + bv.y;
    o4.z = d2 * inv * gv.z + bv.z;
    o4.w = d3 * inv * gv.w + bv.w;
    ((float4*)(y + (size_t)row * H_))[lane] = o4;
}

// Tiny attention (seq <= 4): thread per (b, head, sq).
__global__ void attn_kernel(const float* __restrict__ q, const float* __restrict__ k,
                            const float* __restrict__ v, const float* __restrict__ padj,
                            float* __restrict__ o, int Bb, int S) {
    int t = blockIdx.x * blockDim.x + threadIdx.x;
    if (t >= Bb * NHEAD_ * S) return;
    int sq = t % S;
    int hh = (t / S) % NHEAD_;
    int b  = t / (S * NHEAD_);
    const float* qp = q + (size_t)(b * S + sq) * H_ + hh * DH_;
    float sc[4];
    float mx = -1e30f;
    for (int sk = 0; sk < S; ++sk) {
        const float* kp = k + (size_t)(b * S + sk) * H_ + hh * DH_;
        float d = 0.f;
        for (int e = 0; e < DH_; ++e) d += qp[e] * kp[e];
        d *= 0.17677669529663687f;               // 1/sqrt(32)
        if (padj) d += padj[(size_t)b * S * S + sq * S + sk];
        sc[sk] = d;
        mx = fmaxf(mx, d);
    }
    float sum = 0.f;
    for (int sk = 0; sk < S; ++sk) { sc[sk] = __expf(sc[sk] - mx); sum += sc[sk]; }
    float inv = 1.f / sum;
    float* op = o + (size_t)(b * S + sq) * H_ + hh * DH_;
    for (int e = 0; e < DH_; ++e) {
        float a = 0.f;
        for (int sk = 0; sk < S; ++sk) a += sc[sk] * v[(size_t)(b * S + sk) * H_ + hh * DH_ + e];
        op[e] = a * inv;
    }
}

__global__ void make_idx_kernel(int* __restrict__ out, const int* __restrict__ tgt,
                                int Bb, int T, int NP) {
    int t = blockIdx.x * blockDim.x + threadIdx.x;
    if (t >= Bb * T) return;
    out[t] = tgt[t] + (t / T) * NP;
}

// padj[b,i,j] = adj[b, tgt[b,j], tgt[b,i]]  (reference broadcast semantics)
__global__ void padj_kernel(float* __restrict__ out, const float* __restrict__ adj,
                            const int* __restrict__ tgt, int Bb, int T, int NP) {
    int t = blockIdx.x * blockDim.x + threadIdx.x;
    if (t >= Bb * T * T) return;
    int j = t % T;
    int i = (t / T) % T;
    int b = t / (T * T);
    out[t] = adj[(size_t)b * NP * NP + (size_t)tgt[b * T + j] * NP + tgt[b * T + i]];
}

__global__ void bcast_add_kernel(float* __restrict__ out, const float* __restrict__ cx,
                                 const float* __restrict__ pe, int Bb, int T) {
    int t = blockIdx.x * blockDim.x + threadIdx.x;
    if (t >= Bb * T * H_) return;
    int c = t % H_;
    int b = (t / H_) / T;
    out[t] = cx[(size_t)b * H_ + c] + pe[t];
}

// ---------------- host-side helpers ----------------
static void gemm(hipStream_t st, const float* A, const float* A2, const int* gidx,
                 const float* W, const float* bias, const float* resid, const int* ridx,
                 float* C, int M, int N, int K, int act) {
    dim3 g(ceil_div(M, MT_ * 16), ceil_div(N, 128));
    wmma_gemm_kernel<<<g, 256, 0, st>>>(A, A2, gidx, W, bias, resid, ridx, C, M, N, K, act);
}
static void zero_f(hipStream_t st, float* p, int n) {
    zero_kernel<<<ceil_div(n, 256), 256, 0, st>>>(p, n);
}

struct Blk {
    const float *W1, *W2, *Wk, *Wo, *Wq, *Wv, *b1, *b2, *g1, *be1, *g2, *be2;
};
struct EncScratch { float *h, *q, *k, *v, *att, *f; };

static void run_encoder(hipStream_t st, float* x, int M, int S, const float* padj,
                        const Blk* blks, int nblk, const EncScratch& sc) {
    int Bb = M / S;
    for (int i = 0; i < nblk; ++i) {
        const Blk& p = blks[i];
        ln_kernel<<<ceil_div(M, 8), 256, 0, st>>>(x, p.g1, p.be1, sc.h, M);
        gemm(st, sc.h, nullptr, nullptr, p.Wq, nullptr, nullptr, nullptr, sc.q, M, H_, H_, 0);
        gemm(st, sc.h, nullptr, nullptr, p.Wk, nullptr, nullptr, nullptr, sc.k, M, H_, H_, 0);
        gemm(st, sc.h, nullptr, nullptr, p.Wv, nullptr, nullptr, nullptr, sc.v, M, H_, H_, 0);
        attn_kernel<<<ceil_div(Bb * NHEAD_ * S, 256), 256, 0, st>>>(sc.q, sc.k, sc.v, padj, sc.att, Bb, S);
        gemm(st, sc.att, nullptr, nullptr, p.Wo, nullptr, x, nullptr, x, M, H_, H_, 0);   // x += att@Wo
        ln_kernel<<<ceil_div(M, 8), 256, 0, st>>>(x, p.g2, p.be2, sc.h, M);
        gemm(st, sc.h, nullptr, nullptr, p.W1, p.b1, nullptr, nullptr, sc.f, M, 4 * H_, H_, 1);
        gemm(st, sc.f, nullptr, nullptr, p.W2, p.b2, x, nullptr, x, M, H_, 4 * H_, 0);    // x += ffn
    }
}

// =====================================================================
extern "C" void kernel_launch(void* const* d_in, const int* in_sizes, int n_in,
                              void* d_out, int out_size, void* d_ws, size_t ws_size,
                              hipStream_t stream)
{
    (void)in_sizes; (void)n_in; (void)out_size; (void)ws_size;

    // ---- inputs (setup_inputs dict order) ----
    const float* x_node    = (const float*)d_in[0];
    const float* edge_attr = (const float*)d_in[1];
    const float* rw        = (const float*)d_in[2];
    const int*   nmap      = (const int*)d_in[3];
    const int*   ei        = (const int*)d_in[4];   // (2, ESUB)
    const int*   emap      = (const int*)d_in[5];
    const int*   batch     = (const int*)d_in[6];
    const int*   f2m       = (const int*)d_in[7];
    const int*   m2c       = (const int*)d_in[8];
    const int*   ctx0      = (const int*)d_in[9];
    const int*   tgt0      = (const int*)d_in[10];
    const int*   ctx1      = (const int*)d_in[11];
    const int*   tgt1      = (const int*)d_in[12];
    const int*   tgt2      = (const int*)d_in[13];
    const float* adj0      = (const float*)d_in[14];
    const float* adj1      = (const float*)d_in[15];
    const float* adj2      = (const float*)d_in[16];
    // d_in[17], d_in[18]: masks (unused by reference forward)

    // ---- params: JAX pytree flatten (sorted dict keys), base index 19 ----
    auto Fp = [&](int i) { return (const float*)d_in[i]; };
    const int PB = 19;
    const float* Uw[3]   = { Fp(PB + 0), Fp(PB + 2), Fp(PB + 4) };
    const float* Ubb[3]  = { Fp(PB + 1), Fp(PB + 3), Fp(PB + 5) };
    const float* edgeW = Fp(PB + 6); const float* edgeB = Fp(PB + 7);
    const int BASE_CTX0 = PB + 8, BASE_CTX1 = PB + 32;
    const int BASE_TGT0 = PB + 56, BASE_TGT1 = PB + 80, BASE_TGT2 = PB + 104;
    const float* gnnW[4] = { Fp(PB + 128), Fp(PB + 130), Fp(PB + 132), Fp(PB + 134) };
    const float* gnnB[4] = { Fp(PB + 129), Fp(PB + 131), Fp(PB + 133), Fp(PB + 135) };
    const float* inW = Fp(PB + 136); const float* inB = Fp(PB + 137);
    const float* p00W[3] = { Fp(PB + 138), Fp(PB + 140), Fp(PB + 142) };
    const float* p00B[3] = { Fp(PB + 139), Fp(PB + 141), Fp(PB + 143) };
    const float* p01W[2] = { Fp(PB + 144), Fp(PB + 146) };
    const float* p01B[2] = { Fp(PB + 145), Fp(PB + 147) };
    const float* p12W[2] = { Fp(PB + 148), Fp(PB + 150) };
    const float* p12B[2] = { Fp(PB + 149), Fp(PB + 151) };
    const float* prwW[3] = { Fp(PB + 152), Fp(PB + 154), Fp(PB + 156) };
    const float* prwB[3] = { Fp(PB + 153), Fp(PB + 155), Fp(PB + 157) };

    auto getBlk = [&](int b0) {
        Blk p;
        p.W1 = Fp(b0 + 0); p.W2 = Fp(b0 + 1); p.Wk = Fp(b0 + 2);  p.Wo  = Fp(b0 + 3);
        p.Wq = Fp(b0 + 4); p.Wv = Fp(b0 + 5); p.b1 = Fp(b0 + 6);  p.b2  = Fp(b0 + 7);
        p.g1 = Fp(b0 + 8); p.be1 = Fp(b0 + 9); p.g2 = Fp(b0 + 10); p.be2 = Fp(b0 + 11);
        return p;
    };
    Blk ec0[2] = { getBlk(BASE_CTX0), getBlk(BASE_CTX0 + 12) };
    Blk ec1[2] = { getBlk(BASE_CTX1), getBlk(BASE_CTX1 + 12) };
    Blk et0[2] = { getBlk(BASE_TGT0), getBlk(BASE_TGT0 + 12) };
    Blk et1[2] = { getBlk(BASE_TGT1), getBlk(BASE_TGT1 + 12) };
    Blk et2[2] = { getBlk(BASE_TGT2), getBlk(BASE_TGT2 + 12) };

    // ---- workspace bump allocator ----
    size_t off = 0;
    auto alloc = [&](size_t nbytes) -> void* {
        void* p = (char*)d_ws + off;
        off += (nbytes + 255) & ~(size_t)255;
        return p;
    };
    float* x       = (float*)alloc((size_t)NSUB_ * H_ * 4);
    float* agg     = (float*)alloc((size_t)NSUB_ * H_ * 4);
    float* eenc    = (float*)alloc((size_t)NEDGE_ * H_ * 4);
    float* nodebuf = (float*)alloc((size_t)NNODE_ * H_ * 4);  // x_enc, later node-mean sums
    float* sumP0   = (float*)alloc((size_t)P0_ * H_ * 4);
    float* u       = (float*)alloc((size_t)P0_ * H_ * 4);
    float* sx0     = (float*)alloc((size_t)P0_ * H_ * 4);
    float* sx1     = (float*)alloc((size_t)P1_ * H_ * 4);
    float* sx2     = (float*)alloc((size_t)P2_ * H_ * 4);
    float* raw0    = (float*)alloc((size_t)P0_ * RW_ * 4);
    float* raw1    = (float*)alloc((size_t)P1_ * RW_ * 4);
    float* raw2    = (float*)alloc((size_t)P2_ * RW_ * 4);
    float* cnt     = (float*)alloc((size_t)NNODE_ * 4);
    EncScratch sc;
    sc.h   = (float*)alloc((size_t)512 * H_ * 4);
    sc.q   = (float*)alloc((size_t)512 * H_ * 4);
    sc.k   = (float*)alloc((size_t)512 * H_ * 4);
    sc.v   = (float*)alloc((size_t)512 * H_ * 4);
    sc.att = (float*)alloc((size_t)512 * H_ * 4);
    sc.f   = (float*)alloc((size_t)512 * 4 * H_ * 4);
    float* pe    = (float*)alloc((size_t)512 * H_ * 4);
    float* pin   = (float*)alloc((size_t)512 * H_ * 4);
    float* bufA  = (float*)alloc((size_t)512 * H_ * 4);
    float* bufB  = (float*)alloc((size_t)512 * H_ * 4);
    float* ctxp  = (float*)alloc((size_t)B_ * H_ * 4);
    float* ctxp1 = (float*)alloc((size_t)B_ * H_ * 4);
    float* padjb0 = (float*)alloc((size_t)B_ * 16 * 4);
    float* padjb1 = (float*)alloc((size_t)B_ * 16 * 4);
    float* padjb2 = (float*)alloc((size_t)B_ * 4);
    int* ci0g = (int*)alloc((size_t)B_ * 4);
    int* ti0g = (int*)alloc((size_t)B_ * 4 * 4);
    int* ci1g = (int*)alloc((size_t)B_ * 4);
    int* ti1g = (int*)alloc((size_t)B_ * 4 * 4);
    int* ti2g = (int*)alloc((size_t)B_ * 4);

    float* outp = (float*)d_out;          // tx0,p0,tx1,p1,tx2,p2 concatenated
    float* tx0o = outp;
    float* p0o  = outp + 65536;
    float* tx1o = outp + 131072;
    float* p1o  = outp + 196608;
    float* tx2o = outp + 262144;
    float* p2o  = outp + 278528;

    // ================= pipeline =================
    // 1. node / edge encoders (WMMA GEMMs)
    gemm(stream, x_node, nullptr, nullptr, inW, inB, nullptr, nullptr, nodebuf, NNODE_, H_, FNODE_, 0);
    gemm(stream, edge_attr, nullptr, nullptr, edgeW, edgeB, nullptr, nullptr, eenc, NEDGE_, H_, FEDGE_, 0);
    gather_rows_kernel<<<ceil_div(NSUB_ * H_, 256), 256, 0, stream>>>(x, nodebuf, nmap, NSUB_, H_);

    // raw0 = seg_max(rw[nmap], batch, P0)  (nonneg values -> uint atomicMax, zero init)
    zero_f(stream, raw0, P0_ * RW_);
    segmax_gather_kernel<<<ceil_div(NSUB_ * RW_, 256), 256, 0, stream>>>(rw, nmap, batch, raw0, NSUB_);

    // 2. GNN stack
    for (int L = 0; L < NL_GNN; ++L) {
        if (L > 0) {
            // x += relu(seg_mean(x,batch)@U + b)[batch]  (MLP hoisted to pooled domain)
            zero_f(stream, sumP0, P0_ * H_);
            zero_f(stream, cnt, P0_);
            seg_accum_kernel<<<ceil_div(NSUB_ * H_, 256), 256, 0, stream>>>(x, batch, sumP0, cnt, NSUB_, H_);
            seg_div_kernel<<<ceil_div(P0_ * H_, 256), 256, 0, stream>>>(sumP0, cnt, P0_, H_);
            gemm(stream, sumP0, nullptr, nullptr, Uw[L - 1], Ubb[L - 1], nullptr, nullptr, u, P0_, H_, H_, 1);
            add_gather_kernel<<<ceil_div(NSUB_ * H_, 256), 256, 0, stream>>>(x, u, batch, NSUB_, H_);
            // x = seg_mean(x, nmap, N_NODE)[nmap]
            zero_f(stream, nodebuf, NNODE_ * H_);
            zero_f(stream, cnt, NNODE_);
            seg_accum_kernel<<<ceil_div(NSUB_ * H_, 256), 256, 0, stream>>>(x, nmap, nodebuf, cnt, NSUB_, H_);
            seg_div_kernel<<<ceil_div(NNODE_ * H_, 256), 256, 0, stream>>>(nodebuf, cnt, NNODE_, H_);
            gather_rows_kernel<<<ceil_div(NSUB_ * H_, 256), 256, 0, stream>>>(x, nodebuf, nmap, NSUB_, H_);
        }
        // GINE: agg = segsum(relu(x[src]+e)); x = x + relu((x+agg)@W + b)
        zero_f(stream, agg, NSUB_ * H_);
        gine_msg_kernel<<<ceil_div(ESUB_, 8), 256, 0, stream>>>(x, eenc, ei, emap, agg, ESUB_);
        gemm(stream, x, agg, nullptr, gnnW[L], gnnB[L], x, nullptr, x, NSUB_, H_, H_, 1);
    }

    // 3. hierarchical pooling
    zero_f(stream, sx0, P0_ * H_); zero_f(stream, cnt, P0_);
    seg_accum_kernel<<<ceil_div(NSUB_ * H_, 256), 256, 0, stream>>>(x, batch, sx0, cnt, NSUB_, H_);
    seg_div_kernel<<<ceil_div(P0_ * H_, 256), 256, 0, stream>>>(sx0, cnt, P0_, H_);

    zero_f(stream, sx1, P1_ * H_); zero_f(stream, cnt, P1_);
    seg_accum_kernel<<<ceil_div(P0_ * H_, 256), 256, 0, stream>>>(sx0, f2m, sx1, cnt, P0_, H_);
    seg_div_kernel<<<ceil_div(P1_ * H_, 256), 256, 0, stream>>>(sx1, cnt, P1_, H_);

    zero_f(stream, raw1, P1_ * RW_); zero_f(stream, cnt, P1_);
    seg_accum_kernel<<<ceil_div(P0_ * RW_, 256), 256, 0, stream>>>(raw0, f2m, raw1, cnt, P0_, RW_);
    seg_div_kernel<<<ceil_div(P1_ * RW_, 256), 256, 0, stream>>>(raw1, cnt, P1_, RW_);

    zero_f(stream, sx2, P2_ * H_); zero_f(stream, cnt, P2_);
    seg_accum_kernel<<<ceil_div(P1_ * H_, 256), 256, 0, stream>>>(sx1, m2c, sx2, cnt, P1_, H_);
    seg_div_kernel<<<ceil_div(P2_ * H_, 256), 256, 0, stream>>>(sx2, cnt, P2_, H_);

    zero_f(stream, raw2, P2_ * RW_); zero_f(stream, cnt, P2_);
    seg_accum_kernel<<<ceil_div(P1_ * RW_, 256), 256, 0, stream>>>(raw1, m2c, raw2, cnt, P1_, RW_);
    seg_div_kernel<<<ceil_div(P2_ * RW_, 256), 256, 0, stream>>>(raw2, cnt, P2_, RW_);

    // 4. global indices
    make_idx_kernel<<<1, 256, 0, stream>>>(ci0g, ctx0, B_, 1, NP0_);
    make_idx_kernel<<<2, 256, 0, stream>>>(ti0g, tgt0, B_, 4, NP0_);
    make_idx_kernel<<<1, 256, 0, stream>>>(ci1g, ctx1, B_, 1, NP1_);
    make_idx_kernel<<<2, 256, 0, stream>>>(ti1g, tgt1, B_, 4, NP1_);
    make_idx_kernel<<<1, 256, 0, stream>>>(ti2g, tgt2, B_, 1, NP2_);

    // 5. cx0 = encoder(sx0[ci0] + relu(raw0[ci0]@prw0), ctx blocks)
    gemm(stream, raw0, nullptr, ci0g, prwW[0], prwB[0], sx0, ci0g, ctxp, B_, H_, RW_, 1);
    run_encoder(stream, ctxp, B_, 1, nullptr, ec0, 2, sc);

    // 6. level-0 targets
    gemm(stream, raw0, nullptr, ti0g, prwW[0], prwB[0], nullptr, nullptr, pe, B_ * 4, H_, RW_, 1);
    gather_rows_kernel<<<ceil_div(B_ * 4 * H_, 256), 256, 0, stream>>>(tx0o, sx0, ti0g, B_ * 4, H_);
    padj_kernel<<<ceil_div(B_ * 16, 256), 256, 0, stream>>>(padjb0, adj0, tgt0, B_, 4, NP0_);
    run_encoder(stream, tx0o, B_ * 4, 4, padjb0, et0, 2, sc);
    bcast_add_kernel<<<ceil_div(B_ * 4 * H_, 256), 256, 0, stream>>>(pin, ctxp, pe, B_, 4);
    gemm(stream, pin,  nullptr, nullptr, p00W[0], p00B[0], nullptr, nullptr, bufA, B_ * 4, H_, H_, 1);
    gemm(stream, bufA, nullptr, nullptr, p00W[1], p00B[1], nullptr, nullptr, bufB, B_ * 4, H_, H_, 1);
    gemm(stream, bufB, nullptr, nullptr, p00W[2], p00B[2], nullptr, nullptr, p0o,  B_ * 4, H_, H_, 0);

    // 7. level-1 targets (predicted from cx0)
    gemm(stream, raw1, nullptr, ti1g, prwW[1], prwB[1], nullptr, nullptr, pe, B_ * 4, H_, RW_, 1);
    gather_rows_kernel<<<ceil_div(B_ * 4 * H_, 256), 256, 0, stream>>>(tx1o, sx1, ti1g, B_ * 4, H_);
    padj_kernel<<<ceil_div(B_ * 16, 256), 256, 0, stream>>>(padjb1, adj1, tgt1, B_, 4, NP1_);
    run_encoder(stream, tx1o, B_ * 4, 4, padjb1, et1, 2, sc);
    bcast_add_kernel<<<ceil_div(B_ * 4 * H_, 256), 256, 0, stream>>>(pin, ctxp, pe, B_, 4);
    gemm(stream, pin,  nullptr, nullptr, p01W[0], p01B[0], nullptr, nullptr, bufA, B_ * 4, H_, H_, 1);
    gemm(stream, bufA, nullptr, nullptr, p01W[1], p01B[1], nullptr, nullptr, p1o,  B_ * 4, H_, H_, 0);

    // 8. cx1
    gemm(stream, raw1, nullptr, ci1g, prwW[1], prwB[1], sx1, ci1g, ctxp1, B_, H_, RW_, 1);
    run_encoder(stream, ctxp1, B_, 1, nullptr, ec1, 2, sc);

    // 9. level-2 targets (predicted from cx1)
    gemm(stream, raw2, nullptr, ti2g, prwW[2], prwB[2], nullptr, nullptr, pe, B_, H_, RW_, 1);
    gather_rows_kernel<<<ceil_div(B_ * H_, 256), 256, 0, stream>>>(tx2o, sx2, ti2g, B_, H_);
    padj_kernel<<<1, 256, 0, stream>>>(padjb2, adj2, tgt2, B_, 1, NP2_);
    run_encoder(stream, tx2o, B_, 1, padjb2, et2, 2, sc);
    bcast_add_kernel<<<ceil_div(B_ * H_, 256), 256, 0, stream>>>(pin, ctxp1, pe, B_, 1);
    gemm(stream, pin,  nullptr, nullptr, p12W[0], p12B[0], nullptr, nullptr, bufA, B_, H_, H_, 1);
    gemm(stream, bufA, nullptr, nullptr, p12W[1], p12B[1], nullptr, nullptr, p2o,  B_, H_, H_, 0);
}
